// Pointnet2_52750788329671
// MI455X (gfx1250) — compile-verified, matrix operand order checked by reading, not run on hardware
//
#include <hip/hip_runtime.h>
#include <hip/hip_bf16.h>

#define BN_EPS_F 0.001f

typedef __attribute__((ext_vector_type(16))) _Float16 v16h;
typedef __attribute__((ext_vector_type(8)))  float    v8f;

// CDNA5 WMMA 16-bit fragment: lane L element E holds K = kmap(E, L>=16)
// lanes 0-15 : elems 0-7 -> K 0-7,  elems 8-15 -> K 16-23
// lanes 16-31: elems 0-7 -> K 8-15, elems 8-15 -> K 24-31
__device__ __forceinline__ int wmma_kmap(int e, int hi) {
    return ((e < 8) ? e : (e + 8)) + (hi ? 8 : 0);
}

// Inverse map: packed slot (f16 index) for matrix element (row-or-col r, k).
// Packed layout: [tile16][kstep][lane 0..31][elem 0..15], k32 = Kpad/32.
__device__ __forceinline__ int frag_slot(int r, int k, int k32) {
    int kr = k & 31;
    int hi = (kr >> 3) & 1;
    int e  = (kr & 7) + ((kr >> 4) << 3);
    int lane = (r & 15) + (hi << 4);
    return (((r >> 4) * k32 + (k >> 5)) * 32 + lane) * 16 + e;
}

struct SAParams {
    const _Float16* Wp[3];   // packed weights (fragment order, K padded)
    const float* G[3];
    const float* B[3];
    int C[3];
    int CinPad;              // padded input channels (multiple of 32)
};

// ---------------------------------------------------------------------------
// Pack weight (Kreal x C, f32 row-major) -> fragment-ordered f16, K padded.
// ---------------------------------------------------------------------------
__global__ void pack_w_kernel(const float* __restrict__ W, int Kreal, int Kpad,
                              int C, _Float16* __restrict__ out) {
    int p = blockIdx.x * 256 + threadIdx.x;
    int total = Kpad * C;
    if (p >= total) return;
    int k32 = Kpad >> 5;
    int e = p & 15;
    int lane = (p >> 4) & 31;
    int rest = p >> 9;
    int kstep = rest % k32;
    int ntile = rest / k32;
    int k = kstep * 32 + wmma_kmap(e, lane >> 4);
    int n = ntile * 16 + (lane & 15);
    float v = (k < Kreal) ? W[k * C + n] : 0.0f;
    out[p] = (_Float16)v;
}

// ---------------------------------------------------------------------------
// Pack activations (M x K f32 row-major, K multiple of 32) -> fragment f16.
// ---------------------------------------------------------------------------
__global__ void pack_a_kernel(const float* __restrict__ X, int K,
                              _Float16* __restrict__ out, int total) {
    int p = blockIdx.x * 256 + threadIdx.x;
    if (p >= total) return;
    int k32 = K >> 5;
    int e = p & 15;
    int lane = (p >> 4) & 31;
    int rest = p >> 9;
    int kstep = rest % k32;
    int mtile = rest / k32;
    int k = kstep * 32 + wmma_kmap(e, lane >> 4);
    int row = mtile * 16 + (lane & 15);
    out[p] = (_Float16)X[row * K + k];
}

// ---------------------------------------------------------------------------
// Farthest point sampling: one block per batch, mind[] in dynamic LDS.
// ---------------------------------------------------------------------------
__global__ void fps_kernel(const float* __restrict__ xyz, int N, int Q,
                           int* __restrict__ out) {
    extern __shared__ float mind[];
    __shared__ float rv[256];
    __shared__ int   ri[256];
    const int b = blockIdx.x;
    const int tid = threadIdx.x;
    const int base = b * N * 3;

    for (int j = tid; j < N; j += 256) mind[j] = 1e10f;
    if (tid == 0) out[b * Q] = 0;
    int last = 0;
    __syncthreads();

    for (int it = 1; it < Q; ++it) {
        const float lx = xyz[base + last * 3 + 0];
        const float ly = xyz[base + last * 3 + 1];
        const float lz = xyz[base + last * 3 + 2];
        float bv = -1.0f; int bi = 0;
        for (int j = tid; j < N; j += 256) {
            float dx = xyz[base + j * 3 + 0] - lx;
            float dy = xyz[base + j * 3 + 1] - ly;
            float dz = xyz[base + j * 3 + 2] - lz;
            float d = dx * dx + dy * dy + dz * dz;
            float m = fminf(mind[j], d);
            mind[j] = m;
            if (m > bv) { bv = m; bi = j; }
        }
        rv[tid] = bv; ri[tid] = bi;
        __syncthreads();
        for (int s = 128; s > 0; s >>= 1) {
            if (tid < s) {
                if (rv[tid + s] > rv[tid] ||
                    (rv[tid + s] == rv[tid] && ri[tid + s] < ri[tid])) {
                    rv[tid] = rv[tid + s]; ri[tid] = ri[tid + s];
                }
            }
            __syncthreads();
        }
        last = ri[0];
        if (tid == 0) out[b * Q + it] = last;
        __syncthreads();
    }
}

// ---------------------------------------------------------------------------
// Fused SA: ball query + gather + 3x(WMMA GEMM + BN + ReLU) + max-pool.
// One 256-thread block (8 wave32) per (q, b). Activations live in LDS in
// WMMA fragment order; A = ds_load_b128 pairs, B = global_load_b128 pairs.
// Dynamic LDS: [256B idx][64xCinPad f16 bufA][64x512 f16 bufB]
// ---------------------------------------------------------------------------
__global__ void sa_fused_kernel(const float* __restrict__ xyz, int N,
                                const float* __restrict__ feat, int Cf,
                                const int* __restrict__ fpsIdx, int Q, float r2,
                                SAParams p,
                                float* __restrict__ newXyz,
                                float* __restrict__ outFeat) {
    extern __shared__ char smem[];
    int*      shIdx = (int*)smem;
    _Float16* bufA  = (_Float16*)(smem + 256);
    _Float16* bufB  = (_Float16*)(smem + 256 + 64 * 288 * 2);

    const int tid  = threadIdx.x;
    const int lane = tid & 31;
    const int wave = tid >> 5;
    const int hi   = lane >> 4;
    const int b = blockIdx.y;
    const int q = blockIdx.x;
    const int xbase = b * N * 3;

    const int center = fpsIdx[b * Q + q];
    const float cx = xyz[xbase + center * 3 + 0];
    const float cy = xyz[xbase + center * 3 + 1];
    const float cz = xyz[xbase + center * 3 + 2];

    // ---- ball query by wave 0: first 64 indices (ascending) within radius ----
    if (wave == 0) {
        int cnt = 0;
        for (int base = 0; base < N && cnt < 64; base += 32) {
            int j = base + lane;
            bool inball = false;
            if (j < N) {
                float dx = xyz[xbase + j * 3 + 0] - cx;
                float dy = xyz[xbase + j * 3 + 1] - cy;
                float dz = xyz[xbase + j * 3 + 2] - cz;
                inball = (dx * dx + dy * dy + dz * dz) < r2;
            }
            unsigned long long m = __ballot(inball);
            if (inball) {
                int pos = cnt + (int)__popcll(m & ((1ull << lane) - 1ull));
                if (pos < 64) shIdx[pos] = j;
            }
            cnt += (int)__popcll(m);
        }
        int actual = cnt < 64 ? cnt : 64;
        int fill = (actual > 0) ? shIdx[0] : (N - 1);
        for (int s = actual + lane; s < 64; s += 32) shIdx[s] = fill;
    }
    __syncthreads();

    // ---- gather grouped tile into LDS (fragment order), zero-padded ----
    const int CinPad = p.CinPad;
    const int k32in = CinPad >> 5;
    for (int e = tid; e < 64 * CinPad; e += 256) {
        int s = e / CinPad;
        int c = e - s * CinPad;
        int j = shIdx[s];
        float v = 0.0f;
        if (c < 3)
            v = xyz[xbase + j * 3 + c] - (c == 0 ? cx : (c == 1 ? cy : cz));
        else if (c - 3 < Cf)
            v = feat[(b * N + j) * Cf + (c - 3)];
        bufA[frag_slot(s, c, k32in)] = (_Float16)v;
    }
    __syncthreads();

    // ---- 3 MLP layers ----
    _Float16* X = bufA;
    _Float16* Y = bufB;
    int K = CinPad;
    for (int l = 0; l < 3; ++l) {
        const int Cout = p.C[l];
        const _Float16* Wp = p.Wp[l];
        const float* G  = p.G[l];
        const float* Bt = p.B[l];
        const int k32 = K >> 5;
        const int c32 = Cout >> 5;
        const int tilesN = Cout >> 4;
        const int tiles  = 4 * tilesN;    // 64 rows -> 4 M tiles
        for (int t = wave; t < tiles; t += 8) {
            const int tm = t / tilesN, tn = t - tm * tilesN;
            const _Float16* ap = X  + (tm * k32 * 32 + lane) * 16;
            const _Float16* bp = Wp + (tn * k32 * 32 + lane) * 16;
            v8f acc = {};
#pragma unroll 2
            for (int kk = 0; kk < k32; ++kk) {
                v16h a   = *(const v16h*)(ap + kk * 512);
                v16h bfr = *(const v16h*)(bp + kk * 512);
                acc = __builtin_amdgcn_wmma_f32_16x16x32_f16(
                    false, a, false, bfr, (short)0, acc, false, false);
            }
            const int ncol = tn * 16 + (lane & 15);
            const float sc = G[ncol] * rsqrtf(1.0f + BN_EPS_F);
            const float sh = Bt[ncol];
            // scatter into fragment order for the next layer (K' = Cout)
            const int kr = ncol & 31;
            const int ohi = (kr >> 3) & 1;
            const int oe  = (kr & 7) + ((kr >> 4) << 3);
            const int okstep = ncol >> 5;
            const int obase = (tm * c32 + okstep) * 512 + (ohi << 4) * 16 + oe;
#pragma unroll
            for (int v = 0; v < 8; ++v) {
                int mloc = v + (hi << 3);           // row within tile (0..15)
                float y = fmaxf(acc[v] * sc + sh, 0.0f);
                Y[obase + mloc * 16] = (_Float16)y;
            }
        }
        __syncthreads();
        _Float16* tp = X; X = Y; Y = tp;
        K = Cout;
    }

    // ---- max over the 64 samples (X holds packed final activations) ----
    const int Cl = p.C[2];
    const int k32f = Cl >> 5;
    for (int c = tid; c < Cl; c += 256) {
        int kr = c & 31;
        int chi = (kr >> 3) & 1;
        int ce  = (kr & 7) + ((kr >> 4) << 3);
        int cb  = (c >> 5) * 512 + (chi << 4) * 16 + ce;
        float mx = -1e30f;
        for (int s = 0; s < 64; ++s)
            mx = fmaxf(mx, (float)X[(s >> 4) * (k32f * 512) + cb + (s & 15) * 16]);
        outFeat[(b * Q + q) * Cl + c] = mx;
    }
    if (tid < 3)
        newXyz[(b * Q + q) * 3 + tid] = xyz[xbase + center * 3 + tid];
}

// ---------------------------------------------------------------------------
// FP: 3-NN inverse-distance interpolation + concat(points1) -> f32 row-major
// ---------------------------------------------------------------------------
__global__ void fp_interp_kernel(const float* __restrict__ xyz1, int N1,
                                 const float* __restrict__ xyz2, int N2,
                                 const float* __restrict__ pts2, int C2,
                                 const float* __restrict__ pts1, int C1,
                                 float* __restrict__ out, int total) {
    int idx = blockIdx.x * blockDim.x + threadIdx.x;
    if (idx >= total) return;
    int b = idx / N1;
    const float px = xyz1[idx * 3 + 0];
    const float py = xyz1[idx * 3 + 1];
    const float pz = xyz1[idx * 3 + 2];
    float d0 = 1e30f, d1 = 1e30f, d2 = 1e30f;
    int   i0 = 0, i1 = 0, i2 = 0;
    const int x2b = b * N2 * 3;
    for (int j = 0; j < N2; ++j) {
        float dx = xyz2[x2b + j * 3 + 0] - px;
        float dy = xyz2[x2b + j * 3 + 1] - py;
        float dz = xyz2[x2b + j * 3 + 2] - pz;
        float d = dx * dx + dy * dy + dz * dz;
        if (d < d0)      { d2 = d1; i2 = i1; d1 = d0; i1 = i0; d0 = d; i0 = j; }
        else if (d < d1) { d2 = d1; i2 = i1; d1 = d;  i1 = j; }
        else if (d < d2) { d2 = d;  i2 = j; }
    }
    d0 = fmaxf(d0, 1e-10f); d1 = fmaxf(d1, 1e-10f); d2 = fmaxf(d2, 1e-10f);
    float w0 = 1.0f / d0, w1 = 1.0f / d1, w2 = 1.0f / d2;
    float ws = w0 + w1 + w2;
    w0 /= ws; w1 /= ws; w2 /= ws;
    const int Ct = C1 + C2;
    float* o = out + idx * Ct;
    const float* p0 = pts2 + (b * N2 + i0) * C2;
    const float* p1 = pts2 + (b * N2 + i1) * C2;
    const float* p2 = pts2 + (b * N2 + i2) * C2;
    for (int c = 0; c < C2; ++c) o[c] = w0 * p0[c] + w1 * p1[c] + w2 * p2[c];
    for (int c = 0; c < C1; ++c) o[C2 + c] = pts1[idx * C1 + c];
}

// ---------------------------------------------------------------------------
// GEMM on pre-packed fragments + optional BN/ReLU. One wave32 per 16x16 tile.
// outPacked=1: write f16 fragment order (feeds next GEMM); else f32 row-major.
// ---------------------------------------------------------------------------
__global__ void gemm_packed_wmma_kernel(const _Float16* __restrict__ Ap,
                                        const _Float16* __restrict__ Bp,
                                        const float* __restrict__ G,
                                        const float* __restrict__ Bt,
                                        void* __restrict__ Yout,
                                        int K, int C, int relu, int outPacked) {
    const int lane = threadIdx.x;
    const int hi = lane >> 4;
    const int tm = blockIdx.x, tn = blockIdx.y;
    const int k32 = K >> 5;
    const _Float16* ap = Ap + (tm * k32 * 32 + lane) * 16;
    const _Float16* bp = Bp + (tn * k32 * 32 + lane) * 16;
    v8f acc = {};
#pragma unroll 2
    for (int kk = 0; kk < k32; ++kk) {
        __builtin_prefetch(ap + (kk + 4) * 512, 0, 0);
        v16h a   = *(const v16h*)(ap + kk * 512);
        v16h bfr = *(const v16h*)(bp + kk * 512);
        acc = __builtin_amdgcn_wmma_f32_16x16x32_f16(
            false, a, false, bfr, (short)0, acc, false, false);
    }
    const int ncol = tn * 16 + (lane & 15);
    float sc = 1.0f, sh = 0.0f;
    if (G) { sc = G[ncol] * rsqrtf(1.0f + BN_EPS_F); sh = Bt[ncol]; }
    if (outPacked) {
        _Float16* Yp = (_Float16*)Yout;
        const int c32 = C >> 5;
        const int kr = ncol & 31;
        const int ohi = (kr >> 3) & 1;
        const int oe  = (kr & 7) + ((kr >> 4) << 3);
        const int obase = (tm * c32 + (ncol >> 5)) * 512 + (ohi << 4) * 16 + oe;
#pragma unroll
        for (int v = 0; v < 8; ++v) {
            int mloc = v + (hi << 3);
            float y = acc[v] * sc + sh;
            if (relu) y = fmaxf(y, 0.0f);
            Yp[obase + mloc * 16] = (_Float16)y;
        }
    } else {
        float* Yf = (float*)Yout;
#pragma unroll
        for (int v = 0; v < 8; ++v) {
            int m = tm * 16 + v + (hi << 3);
            float y = acc[v] * sc + sh;
            if (relu) y = fmaxf(y, 0.0f);
            Yf[m * C + ncol] = y;
        }
    }
}

// ---------------------------------------------------------------------------
extern "C" void kernel_launch(void* const* d_in, const int* in_sizes, int n_in,
                              void* d_out, int out_size, void* d_ws, size_t ws_size,
                              hipStream_t stream) {
    const float* input = (const float*)d_in[0];   // (8, 8192, 3)

    const float *saW[12], *saG[12], *saB[12];
    const float *fpW[8],  *fpG[8],  *fpB[8];

    if (n_in >= 59 && in_sizes[1] == 96) {
        // dict-insertion order: sa first, each layer (w, gamma, beta)
        int i = 1;
        for (int L = 0; L < 12; ++L) {
            saW[L] = (const float*)d_in[i++];
            saG[L] = (const float*)d_in[i++];
            saB[L] = (const float*)d_in[i++];
        }
        for (int L = 0; L < 8; ++L) {
            fpW[L] = (const float*)d_in[i++];
            if (L == 7) { fpG[L] = nullptr; fpB[L] = nullptr; }
            else { fpG[L] = (const float*)d_in[i++]; fpB[L] = (const float*)d_in[i++]; }
        }
    } else {
        // jax pytree sorted order: fp first, each layer (beta, gamma, w)
        int i = 1;
        for (int L = 0; L < 8; ++L) {
            if (L == 7) { fpB[L] = nullptr; fpG[L] = nullptr; fpW[L] = (const float*)d_in[i++]; }
            else {
                fpB[L] = (const float*)d_in[i++];
                fpG[L] = (const float*)d_in[i++];
                fpW[L] = (const float*)d_in[i++];
            }
        }
        for (int L = 0; L < 12; ++L) {
            saB[L] = (const float*)d_in[i++];
            saG[L] = (const float*)d_in[i++];
            saW[L] = (const float*)d_in[i++];
        }
    }

    // ---- workspace layout (floats, 256B-aligned chunks) ----
    float* wsf = (float*)d_ws;
    size_t off = 0;
    auto alloc = [&](size_t n) -> float* {
        n = (n + 63) & ~(size_t)63;
        float* p = wsf + off; off += n; return p;
    };

    float* xyz1 = alloc(8 * 1024 * 3);
    float* xyz2 = alloc(8 * 256 * 3);
    float* xyz3 = alloc(8 * 64 * 3);
    float* xyz4 = alloc(8 * 16 * 3);
    float* pts1 = alloc(8 * 1024 * 64);
    float* pts2 = alloc(8 * 256 * 128);
    float* pts3 = alloc(8 * 64 * 256);
    float* pts4 = alloc(8 * 16 * 512);
    float* np3  = alloc(8 * 64 * 128);
    float* np2  = alloc(8 * 256 * 128);
    float* np1  = alloc(8 * 1024 * 64);
    int*   fidx = (int*)alloc(8 * 1024);
    float* INb  = alloc((size_t)65536 * 64);              // interp output (f32)
    _Float16* APK = (_Float16*)alloc((size_t)65536 * 32); // packed A (f16)
    _Float16* TPK = (_Float16*)alloc((size_t)65536 * 32); // packed intermediate

    // packed weights
    static const int saKpad[12]  = {32,32,32, 96,64,64, 160,128,128, 288,256,256};
    static const int saKreal[12] = {3,32,32, 67,64,64, 131,128,128, 259,256,256};
    static const int saCout[12]  = {32,32,64, 64,64,128, 128,128,256, 256,256,512};
    static const int fpKpad[8]   = {768,128, 256,128, 192,128, 64,64};
    static const int fpCout[8]   = {128,128, 128,128, 128,64, 64,64};
    _Float16* saWp[12]; _Float16* fpWp[8];
    for (int L = 0; L < 12; ++L) saWp[L] = (_Float16*)alloc((size_t)saKpad[L] * saCout[L] / 2 + 32);
    for (int L = 0; L < 8;  ++L) fpWp[L] = (_Float16*)alloc((size_t)fpKpad[L] * fpCout[L] / 2 + 32);
    if (off * sizeof(float) > ws_size) return;

    for (int L = 0; L < 12; ++L) {
        int total = saKpad[L] * saCout[L];
        pack_w_kernel<<<(total + 255) / 256, 256, 0, stream>>>(
            saW[L], saKreal[L], saKpad[L], saCout[L], saWp[L]);
    }
    for (int L = 0; L < 8; ++L) {
        int total = fpKpad[L] * fpCout[L];
        pack_w_kernel<<<(total + 255) / 256, 256, 0, stream>>>(
            fpW[L], fpKpad[L], fpKpad[L], fpCout[L], fpWp[L]);
    }

    const int B = 8;
    const size_t saLds = 256 + 64 * 288 * 2 + 64 * 512 * 2;  // ~102.7 KB

    struct { const float* xyzIn; const float* featIn; int N, Q, Cf, CinPad;
             float r2; float* xyzOut; float* ptsOut; } lv[4] = {
        { input, nullptr, 8192, 1024, 0,   32,  0.04f * 0.04f, xyz1, pts1 },
        { xyz1,  pts1,    1024, 256,  64,  96,  0.08f * 0.08f, xyz2, pts2 },
        { xyz2,  pts2,    256,  64,   128, 160, 0.16f * 0.16f, xyz3, pts3 },
        { xyz3,  pts3,    64,   16,   256, 288, 0.32f * 0.32f, xyz4, pts4 },
    };

    for (int L = 0; L < 4; ++L) {
        fps_kernel<<<B, 256, (size_t)lv[L].N * sizeof(float), stream>>>(
            lv[L].xyzIn, lv[L].N, lv[L].Q, fidx);
        SAParams p;
        for (int j = 0; j < 3; ++j) {
            p.Wp[j] = saWp[L * 3 + j];
            p.G[j]  = saG[L * 3 + j];
            p.B[j]  = saB[L * 3 + j];
            p.C[j]  = saCout[L * 3 + j];
        }
        p.CinPad = lv[L].CinPad;
        sa_fused_kernel<<<dim3(lv[L].Q, B), 256, saLds, stream>>>(
            lv[L].xyzIn, lv[L].N, lv[L].featIn, lv[L].Cf,
            fidx, lv[L].Q, lv[L].r2, p, lv[L].xyzOut, lv[L].ptsOut);
    }

    struct { const float* x1; int N1; const float* x2; int N2;
             const float* p2src; int C2; const float* p1src; int C1;
             int C0, C1o; float* outp; int lastRelu; } fp[4] = {
        { xyz3,  64,   xyz4, 16,   pts4, 512, pts3,    256, 128, 128, np3, 1 },
        { xyz2,  256,  xyz3, 64,   np3,  128, pts2,    128, 128, 128, np2, 1 },
        { xyz1,  1024, xyz2, 256,  np2,  128, pts1,    64,  128, 64,  np1, 1 },
        { input, 8192, xyz1, 1024, np1,  64,  nullptr, 0,   64,  64,  (float*)d_out, 0 },
    };
    for (int i = 0; i < 4; ++i) {
        int M   = B * fp[i].N1;
        int Kin = fp[i].C1 + fp[i].C2;
        fp_interp_kernel<<<(M + 255) / 256, 256, 0, stream>>>(
            fp[i].x1, fp[i].N1, fp[i].x2, fp[i].N2,
            fp[i].p2src, fp[i].C2, fp[i].p1src, fp[i].C1, INb, M);
        int totalA = M * Kin;
        pack_a_kernel<<<(totalA + 255) / 256, 256, 0, stream>>>(INb, Kin, APK, totalA);
        gemm_packed_wmma_kernel<<<dim3(M / 16, fp[i].C0 / 16), 32, 0, stream>>>(
            APK, fpWp[i * 2], fpG[i * 2], fpB[i * 2], TPK, Kin, fp[i].C0, 1, 1);
        gemm_packed_wmma_kernel<<<dim3(M / 16, fp[i].C1o / 16), 32, 0, stream>>>(
            TPK, fpWp[i * 2 + 1], fpG[i * 2 + 1], fpB[i * 2 + 1], fp[i].outp,
            fp[i].C0, fp[i].C1o, fp[i].lastRelu, 0);
    }
}
